// OptimizedRNN_32701880991972
// MI455X (gfx1250) — matrix operand order repaired and checked by reading, not compiled, not run on hardware
//
#include <hip/hip_runtime.h>
#include <hip/hip_bf16.h>
#include <math.h>

// ---------------------------------------------------------------------------
// MI455X (gfx1250) GELU-RNN:
//   x_proj = seq @ Wi^T + bi            (bf16 WMMA GEMM, M=32768)
//   h_t    = gelu(x_t + h_{t-1} @ Wh^T) (1024 small bf16 WMMA GEMM steps)
//   out    = hidden @ Wo^T + bo         (bf16 WMMA GEMM, M=32768)
// All matmuls use v_wmma_f32_16x16x32_bf16 (f32 accumulate).
// Recurrence state is staged into LDS with GLOBAL_LOAD_ASYNC_TO_LDS_B128.
// ---------------------------------------------------------------------------

typedef __attribute__((ext_vector_type(16))) __bf16 v16bf;
typedef __attribute__((ext_vector_type(8)))  float  v8f;
typedef __attribute__((ext_vector_type(4)))  int    v4i;

#if __has_builtin(__builtin_amdgcn_global_load_async_to_lds_b128) && \
    __has_builtin(__builtin_amdgcn_s_wait_asynccnt)
#define USE_ASYNC_LDS 1
#else
#define USE_ASYNC_LDS 0
#endif

union Frag16 {
  uint4        q[2];   // two 16-byte chunks
  unsigned int u[8];
  v16bf        bf;     // 16 bf16 halves per lane (A or B operand of 16x16x32)
};

__device__ __forceinline__ unsigned short f2bf(float f) {
  unsigned int u = __float_as_uint(f);
  return (unsigned short)((u + 0x7FFFu + ((u >> 16) & 1u)) >> 16);  // RNE
}
__device__ __forceinline__ unsigned int pack_bf16(float lo, float hi) {
  return ((unsigned int)f2bf(hi) << 16) | (unsigned int)f2bf(lo);
}
__device__ __forceinline__ float gelu_exact(float x) {
  return 0.5f * x * (1.0f + erff(x * 0.70710678118654752f));
}

// --------------------------- tiny utility kernels --------------------------

__global__ void cvt_f32_bf16(const float* __restrict__ src,
                             unsigned short* __restrict__ dst, int n) {
  int i = blockIdx.x * blockDim.x + threadIdx.x;
  if (i < n) dst[i] = f2bf(src[i]);
}

__global__ void zero_u32(unsigned int* __restrict__ dst, int n) {
  int i = blockIdx.x * blockDim.x + threadIdx.x;
  if (i < n) dst[i] = 0u;
}

// ------------------------------- big GEMM ----------------------------------
// C[M,N](f32) = A[M,K=1024](f32) @ W[N,K=1024](bf16)^T + bias
// Block tile 128x128, 8 waves (wave32) as 4(M) x 2(N); each wave owns a 32x64
// macro-tile = 2x4 WMMA tiles (64 accumulator VGPRs) -> 8 WMMAs per 32-wide K
// chunk against 2KB LDS + 4KB L2 of operand traffic. A is converted f32->bf16
// while staged into padded LDS; B fragments stream directly from L2 (each
// weight is 2 MB, fully resident in the 192 MB L2).
#define GK 1024

__global__ __launch_bounds__(256)
void gemm_bias(const float* __restrict__ A, const unsigned short* __restrict__ Wb,
               const float* __restrict__ bias, float* __restrict__ C, int Nld) {
  __shared__ unsigned short As[128 * 72];   // 72 = 64 + 8 pad halves -> 144B rows

  const int tid  = threadIdx.x;
  const int lane = tid & 31;
  const int wave = tid >> 5;
  const int wm   = wave >> 1;       // 4 M wave-groups (32 rows each)
  const int wn   = wave & 1;        // 2 N wave-groups (64 cols each)
  const int r    = lane & 15;
  const int hi   = lane >> 4;
  const int mbase = blockIdx.y * 128;
  const int nbase = blockIdx.x * 128;

  v8f acc[2][4] = {};

  // staging: 2 threads per row, 32 consecutive halves each
  const int srow = tid >> 1;
  const int scol = (tid & 1) * 32;
  const float* aptr = A + (size_t)(mbase + srow) * GK + scol;
  unsigned short* sptr = &As[srow * 72 + scol];

  for (int kc = 0; kc < GK; kc += 64) {
    __syncthreads();
    const float4* a4 = (const float4*)(aptr + kc);
    if (kc + 64 < GK)
      __builtin_prefetch(aptr + kc + 64, 0, 0);   // global_prefetch_b8
#pragma unroll
    for (int i = 0; i < 4; ++i) {
      float4 x0 = a4[2 * i];
      float4 x1 = a4[2 * i + 1];
      uint4 p;
      p.x = pack_bf16(x0.x, x0.y);
      p.y = pack_bf16(x0.z, x0.w);
      p.z = pack_bf16(x1.x, x1.y);
      p.w = pack_bf16(x1.z, x1.w);
      *(uint4*)(sptr + i * 8) = p;
    }
    __syncthreads();

#pragma unroll
    for (int kk = 0; kk < 64; kk += 32) {
      Frag16 a[2], b[4];
#pragma unroll
      for (int i = 0; i < 2; ++i) {
        // A 16x32 layout: lane r row M, halves = K {16*hi .. +16} contiguous
        const unsigned short* ab = &As[((wm * 2 + i) * 16 + r) * 72 + kk + 16 * hi];
        a[i].q[0] = *(const uint4*)(ab);
        a[i].q[1] = *(const uint4*)(ab + 8);
      }
#pragma unroll
      for (int j = 0; j < 4; ++j) {
        // B 32x16 layout: col = lane&15, halves = K {16*hi .. +16} contiguous
        int n = nbase + (wn * 4 + j) * 16 + r;
        const unsigned short* gb = Wb + (size_t)n * GK + kc + kk + 16 * hi;
        b[j].q[0] = *(const uint4*)(gb);
        b[j].q[1] = *(const uint4*)(gb + 8);
      }
#pragma unroll
      for (int i = 0; i < 2; ++i)
#pragma unroll
        for (int j = 0; j < 4; ++j)
          acc[i][j] = __builtin_amdgcn_wmma_f32_16x16x32_bf16(
              false, a[i].bf, false, b[j].bf, (short)0, acc[i][j], false, false);
    }
  }

  // C layout: M = vgpr + 8*hi, N = lane&15
#pragma unroll
  for (int i = 0; i < 2; ++i) {
    int row0 = mbase + (wm * 2 + i) * 16 + 8 * hi;
#pragma unroll
    for (int j = 0; j < 4; ++j) {
      int col = nbase + (wn * 4 + j) * 16 + r;
      float bv = bias[col];
#pragma unroll
      for (int v = 0; v < 8; ++v)
        C[(size_t)(row0 + v) * Nld + col] = acc[i][j][v] + bv;
    }
  }
}

// ---------------------------- recurrence step ------------------------------
// h_new[32,1024] = gelu(x_t + h_prev @ Wh^T); h carried in bf16, gelu in f32.
// 16 WGs x 256 threads; per WG a 64-column slice; 8 waves = 2(M) x 4(N), one
// 16x16 C tile per wave. h (32xK) staged into LDS in 128-wide K chunks with
// GLOBAL_LOAD_ASYNC_TO_LDS_B128 (ASYNCcnt) when available.
__global__ __launch_bounds__(256)
void rnn_step(const float* __restrict__ xp,            // + t*H, row stride T*H
              const unsigned short* __restrict__ hin,  // bf16 32x1024
              const unsigned short* __restrict__ Whb,  // bf16 1024x1024 (N,K)
              float* __restrict__ hout,                // + t*H, row stride T*H
              unsigned short* __restrict__ hout_bf) {  // bf16 32x1024
  __shared__ unsigned short Hs[32 * 136];   // 136 = 128 + 8 pad halves

  const int tid  = threadIdx.x;
  const int lane = tid & 31;
  const int wave = tid >> 5;
  const int wm   = wave & 1;
  const int wn   = wave >> 1;               // 0..3
  const int r    = lane & 15;
  const int hi   = lane >> 4;
  const int nbase = blockIdx.x * 64;
  const size_t rs = (size_t)1024 * 1024;    // T*H row stride

  v8f acc = {};

  // staging: 8 threads per row, 16 halves (32B) each
  const int srow = tid >> 3;
  const int sseg = (tid & 7) * 16;

  for (int kc = 0; kc < 1024; kc += 128) {
    __syncthreads();
    const unsigned short* g0 = hin + srow * 1024 + kc + sseg;
    unsigned short* s0 = &Hs[srow * 136 + sseg];
#if USE_ASYNC_LDS
    // direct memory->LDS DMA (ASYNCcnt), no VGPR round trip
    __builtin_amdgcn_global_load_async_to_lds_b128((v4i*)g0, (v4i*)s0, 0, 0);
    __builtin_amdgcn_global_load_async_to_lds_b128((v4i*)(g0 + 8), (v4i*)(s0 + 8), 0, 0);
    __builtin_amdgcn_s_wait_asynccnt(0);
#else
    {
      const uint4* g = (const uint4*)g0;
      uint4* s = (uint4*)s0;
      s[0] = g[0];
      s[1] = g[1];
    }
#endif
    __syncthreads();
#pragma unroll
    for (int kk = 0; kk < 128; kk += 32) {
      Frag16 a, b;
      const unsigned short* ab = &Hs[(wm * 16 + r) * 136 + kk + 16 * hi];
      a.q[0] = *(const uint4*)(ab);
      a.q[1] = *(const uint4*)(ab + 8);
      int n = nbase + wn * 16 + r;
      const unsigned short* gb = Whb + (size_t)n * 1024 + kc + kk + 16 * hi;
      b.q[0] = *(const uint4*)(gb);
      b.q[1] = *(const uint4*)(gb + 8);
      acc = __builtin_amdgcn_wmma_f32_16x16x32_bf16(
          false, a.bf, false, b.bf, (short)0, acc, false, false);
    }
  }

  int col  = nbase + wn * 16 + r;
  int row0 = wm * 16 + 8 * hi;
#pragma unroll
  for (int v = 0; v < 8; ++v) {
    int row = row0 + v;
    float x = acc[v] + xp[(size_t)row * rs + col];
    float gv = gelu_exact(x);
    hout[(size_t)row * rs + col] = gv;
    hout_bf[row * 1024 + col] = f2bf(gv);
  }
}

// ------------------------------- launcher ----------------------------------

extern "C" void kernel_launch(void* const* d_in, const int* in_sizes, int n_in,
                              void* d_out, int out_size, void* d_ws, size_t ws_size,
                              hipStream_t stream) {
  (void)in_sizes; (void)n_in; (void)out_size; (void)ws_size;
  const float* seq = (const float*)d_in[0];   // (32,1024,1024)
  const float* Wi  = (const float*)d_in[1];   // (1024,1024)
  const float* bi  = (const float*)d_in[2];   // (1024,)
  const float* Wh  = (const float*)d_in[3];   // (1024,1024)
  const float* Wo  = (const float*)d_in[4];   // (1024,1024)
  const float* bo  = (const float*)d_in[5];   // (1024,)

  const int B = 32, T = 1024, H = 1024, D = 1024;
  const size_t MH = (size_t)B * T * H;

  float* hidden = (float*)d_out;              // outputs[0]
  float* output = (float*)d_out + MH;         // outputs[1]; also x_proj scratch

  // workspace: bf16 weights + bf16 h ping-pong (~6.3 MB)
  unsigned short* Wib = (unsigned short*)d_ws;
  unsigned short* Whb = Wib + (size_t)H * D;
  unsigned short* Wob = Whb + (size_t)H * H;
  unsigned short* h0  = Wob + (size_t)H * H;
  unsigned short* h1  = h0 + (size_t)B * H;

  const int wcount = H * D;  // 1M elements per weight
  cvt_f32_bf16<<<dim3((wcount + 255) / 256), 256, 0, stream>>>(Wi, Wib, wcount);
  cvt_f32_bf16<<<dim3((wcount + 255) / 256), 256, 0, stream>>>(Wh, Whb, wcount);
  cvt_f32_bf16<<<dim3((wcount + 255) / 256), 256, 0, stream>>>(Wo, Wob, wcount);
  const int hwords = (B * H) / 2;  // bf16 pairs as u32
  zero_u32<<<dim3((hwords + 255) / 256), 256, 0, stream>>>((unsigned int*)h0, hwords);

  // input projection: x_proj -> output region (consumed by recurrence before
  // the final GEMM overwrites it)
  gemm_bias<<<dim3(H / 128, (B * T) / 128), 256, 0, stream>>>(seq, Wib, bi, output, H);

  // 1024 sequential recurrence steps (kernel boundary = device-wide sync)
  unsigned short* hp = h0;
  unsigned short* hn = h1;
  for (int t = 0; t < T; ++t) {
    rnn_step<<<dim3(16), 256, 0, stream>>>(output + (size_t)t * H, hp, Whb,
                                           hidden + (size_t)t * H, hn);
    unsigned short* tmp = hp; hp = hn; hn = tmp;
  }

  // output projection from f32 hidden states
  gemm_bias<<<dim3(H / 128, (B * T) / 128), 256, 0, stream>>>(hidden, Wob, bo, output, H);
}
